// AlterSRegressor_89507118449122
// MI455X (gfx1250) — matrix operand order repaired and checked by reading, not compiled, not run on hardware
//
#include <hip/hip_runtime.h>
#include <cstdint>

// ---------------- problem constants (match reference) ----------------
#define G_    32
#define NPER_ 512
#define N_    (G_ * NPER_)      // 16384
#define DEG_  32
#define E_    (N_ * DEG_)       // 524288
#define IN_   64
#define EIN_  16
#define HID_  128
#define H_    4
#define C_    32
#define LR_   32

typedef __attribute__((ext_vector_type(16))) _Float16 v16h;
typedef __attribute__((ext_vector_type(8)))  _Float16 v8h;
typedef __attribute__((ext_vector_type(8)))  float    v8f;

union AV { v16h v; v8h p[2]; };

__device__ __forceinline__ float wave_sum32(float t) {
  for (int o = 16; o; o >>= 1) t += __shfl_xor(t, o, 32);
  return t;
}

// ---------------- generic batched WMMA GEMM, 32x32 register tile ----------------
// C[M][ldc] = A[M][lda](f16) * B + bias, where B is given transposed:
// BT[Ncols][ldb](f16), BT[n][k] = B[k][n].  Kd multiple of 32; M, Ncols mult of 32.
// Each wave owns a 32x32 output tile = 2x2 grid of 16x16 WMMA tiles: per K-step
// it loads 2 A fragments + 2 B fragments and issues 4 independent WMMAs
// (16 FLOP/byte vs 8 for a single-tile wave; everything is L2-resident).
__global__ void wmma_gemm_kernel(const _Float16* __restrict__ A,
                                 const _Float16* __restrict__ BT,
                                 const float* __restrict__ bias,
                                 float* __restrict__ C,
                                 int M, int Ncols, int Kd,
                                 int lda, int ldb, int ldc,
                                 long long batchA, long long batchB, long long batchC) {
  const int wave = threadIdx.x >> 5;
  const int lane = threadIdx.x & 31;
  const int tile = blockIdx.x * (blockDim.x >> 5) + wave;
  const int tilesN = Ncols >> 5;
  const int tilesM = M >> 5;
  if (tile >= tilesM * tilesN) return;   // uniform per wave (EXEC stays all-1s)

  A  += (long long)blockIdx.y * batchA;
  BT += (long long)blockIdx.y * batchB;
  C  += (long long)blockIdx.y * batchC;

  const int tm = (tile / tilesN) << 5;
  const int tn = (tile % tilesN) << 5;
  const int lr = lane & 15;
  const int aoff = (lane < 16) ? 0 : 8;    // A: lanes0-15 K={0..7,16..23}, lanes16-31 K={8..15,24..31}
  const int koff = (lane < 16) ? 0 : 8;    // B: lanes0-15 K={0..7,16..23}, lanes16-31 K={8..15,24..31}

  const _Float16* arow0 = A + (size_t)(tm + lr) * lda + aoff;
  const _Float16* arow1 = arow0 + (size_t)16 * lda;
  const _Float16* brow0 = BT + (size_t)(tn + lr) * ldb + koff;
  const _Float16* brow1 = brow0 + (size_t)16 * ldb;

  v8f acc00 = {}, acc01 = {}, acc10 = {}, acc11 = {};
  for (int kb = 0; kb < Kd; kb += 32) {
    __builtin_prefetch(arow0 + kb + 64, 0, 1);
    __builtin_prefetch(arow1 + kb + 64, 0, 1);
    AV a0, a1;
    a0.p[0] = *(const v8h*)(arow0 + kb);
    a0.p[1] = *(const v8h*)(arow0 + kb + 16);
    a1.p[0] = *(const v8h*)(arow1 + kb);
    a1.p[1] = *(const v8h*)(arow1 + kb + 16);
    AV bb0, bb1;
    bb0.p[0] = *(const v8h*)(brow0 + kb);
    bb0.p[1] = *(const v8h*)(brow0 + kb + 16);
    bb1.p[0] = *(const v8h*)(brow1 + kb);
    bb1.p[1] = *(const v8h*)(brow1 + kb + 16);
    v16h b0 = bb0.v;
    v16h b1 = bb1.v;
    acc00 = __builtin_amdgcn_wmma_f32_16x16x32_f16(false, a0.v, false, b0,
                                                   (short)0, acc00, false, false);
    acc01 = __builtin_amdgcn_wmma_f32_16x16x32_f16(false, a0.v, false, b1,
                                                   (short)0, acc01, false, false);
    acc10 = __builtin_amdgcn_wmma_f32_16x16x32_f16(false, a1.v, false, b0,
                                                   (short)0, acc10, false, false);
    acc11 = __builtin_amdgcn_wmma_f32_16x16x32_f16(false, a1.v, false, b1,
                                                   (short)0, acc11, false, false);
  }
  const float bv0 = bias ? bias[tn + lr] : 0.f;
  const float bv1 = bias ? bias[tn + 16 + lr] : 0.f;
  const int rbase = tm + ((lane < 16) ? 0 : 8);
#pragma unroll
  for (int r = 0; r < 8; ++r) {
    float* c0 = C + (size_t)(rbase + r) * ldc;
    float* c1 = C + (size_t)(rbase + 16 + r) * ldc;
    c0[tn + lr]      = acc00[r] + bv0;
    c0[tn + 16 + lr] = acc01[r] + bv1;
    c1[tn + lr]      = acc10[r] + bv0;
    c1[tn + 16 + lr] = acc11[r] + bv1;
  }
}

// ---------------- misc small kernels ----------------
__global__ void zero_u32(unsigned* p, int n) {
  int i = blockIdx.x * blockDim.x + threadIdx.x;
  if (i < n) p[i] = 0u;
}

// f = sigmoid(relu(ea@w1+b1)@w2+b2)
__global__ void edge_mlp(const float* __restrict__ ea, const float* __restrict__ w1,
                         const float* __restrict__ b1, const float* __restrict__ w2,
                         const float* __restrict__ b2, float* __restrict__ f) {
  int e = blockIdx.x * blockDim.x + threadIdx.x;
  if (e >= E_) return;
  float a[EIN_];
#pragma unroll
  for (int i = 0; i < EIN_; ++i) a[i] = ea[(size_t)e * EIN_ + i];
  float acc = b2[0];
  for (int j = 0; j < HID_; ++j) {
    float s = b1[j];
#pragma unroll
    for (int i = 0; i < EIN_; ++i) s += a[i] * w1[i * HID_ + j];
    acc += fmaxf(s, 0.f) * w2[j];
  }
  f[e] = 1.f / (1.f + __expf(-acc));
}

__global__ void scatter_R(const int* __restrict__ src, const int* __restrict__ dst,
                          const float* __restrict__ f, float* __restrict__ R) {
  int e = blockIdx.x * blockDim.x + threadIdx.x;
  if (e >= E_) return;
  int s = src[e], d = dst[e];
  int g = s >> 9;
  atomicAdd(&R[((size_t)g << 18) + ((size_t)(s & 511) << 9) + (size_t)(d & 511)], f[e]);
}

// column sums (axis u), normalize in place, emit f16 copy
__global__ void colnorm_R(float* __restrict__ R, _Float16* __restrict__ Rh) {
  int t = blockIdx.x * blockDim.x + threadIdx.x;   // G*512 columns
  int g = t >> 9, v = t & 511;
  float* Rg = R + ((size_t)g << 18);
  _Float16* Rhg = Rh + ((size_t)g << 18);
  float s = 0.f;
  for (int u = 0; u < NPER_; ++u) s += Rg[((size_t)u << 9) + v];
  float inv = (s == 0.f) ? 1.f : (1.f / s);
  for (int u = 0; u < NPER_; ++u) {
    size_t idx = ((size_t)u << 9) + v;
    float val = Rg[idx] * inv;
    Rg[idx] = val;
    Rhg[idx] = (_Float16)val;
  }
}

// LDS-tiled f16 transpose per graph: out[x][y] = in[y][x]
__global__ void transpose_h16(const _Float16* __restrict__ in, _Float16* __restrict__ out) {
  __shared__ _Float16 tile[32][33];
  int g = blockIdx.z;
  const _Float16* I = in + ((size_t)g << 18);
  _Float16* O = out + ((size_t)g << 18);
  int x0 = blockIdx.x * 32, y0 = blockIdx.y * 32;
  int tx = threadIdx.x, ty = threadIdx.y;     // 32 x 8
#pragma unroll
  for (int i = 0; i < 32; i += 8)
    tile[ty + i][tx] = I[(size_t)(y0 + ty + i) * NPER_ + x0 + tx];
  __syncthreads();
#pragma unroll
  for (int i = 0; i < 32; i += 8)
    O[(size_t)(x0 + ty + i) * NPER_ + y0 + tx] = tile[tx][ty + i];
}

// diag features -> long-range proj -> h0 (cols 0..63 = x, 64..95 = E_b), + f16 copy
__global__ void build_h0(const float* __restrict__ Rn, const _Float16* __restrict__ RhT,
                         const float* __restrict__ P, const float* __restrict__ x,
                         const float* __restrict__ lr_w, const float* __restrict__ lr_b,
                         float* __restrict__ h, _Float16* __restrict__ hh) {
  int wid = threadIdx.x >> 5, lane = threadIdx.x & 31;
  int n = blockIdx.x * 8 + wid;
  int g = n >> 9, l = n & 511;
  const float* Prow = P + ((size_t)g << 18) + ((size_t)l << 9);
  const _Float16* Trow = RhT + ((size_t)g << 18) + ((size_t)l << 9);
  float acc = 0.f;
  for (int j = lane; j < NPER_; j += 32) acc += Prow[j] * (float)Trow[j];
  float d3 = wave_sum32(acc);                     // diag(R^3)
  float d1 = Rn[((size_t)g << 18) + (size_t)l * 513];  // diag(R)
  float d2 = Prow[l];                             // diag(R^2)
  // E_b[lane] = [1,d1,d2,d3] @ lr_w + lr_b
  float eb = lr_b[lane] + lr_w[lane] + d1 * lr_w[32 + lane] +
             d2 * lr_w[64 + lane] + d3 * lr_w[96 + lane];
  size_t base = (size_t)n * HID_;
  float x0 = x[(size_t)n * IN_ + lane];
  float x1 = x[(size_t)n * IN_ + 32 + lane];
  h[base + lane] = x0;            hh[base + lane] = (_Float16)x0;
  h[base + 32 + lane] = x1;       hh[base + 32 + lane] = (_Float16)x1;
  h[base + 64 + lane] = eb;       hh[base + 64 + lane] = (_Float16)eb;
}

// transpose+convert weight w[rows][128](f32) -> wt[128][128](f16), zero-pad rows
__global__ void wT_f16(const float* __restrict__ w, _Float16* __restrict__ wt, int rows) {
  int t = blockIdx.x * blockDim.x + threadIdx.x;  // 128*128
  int c = t >> 7, i = t & 127;
  wt[t] = (i < rows) ? (_Float16)w[i * HID_ + c] : (_Float16)0.f;
}

__device__ __forceinline__ unsigned enc_f32(float f) {
  unsigned u = __float_as_uint(f);
  return (u & 0x80000000u) ? ~u : (u | 0x80000000u);
}
__device__ __forceinline__ float dec_f32(unsigned u) {
  return (u & 0x80000000u) ? __uint_as_float(u ^ 0x80000000u) : __uint_as_float(~u);
}

// pass 1: per-edge per-head logits + segment max (uint-encoded float atomicMax)
__global__ void attn_logits(const float* __restrict__ Q, const float* __restrict__ Kb,
                            const float* __restrict__ we, const float* __restrict__ ea,
                            const int* __restrict__ src, const int* __restrict__ dst,
                            float* __restrict__ logits, unsigned* __restrict__ mU) {
  int e = blockIdx.x, j = threadIdx.x;   // 128 threads: wave == head
  __shared__ float eas[EIN_];
  if (j < EIN_) eas[j] = ea[(size_t)e * EIN_ + j];
  __syncthreads();
  float ej = 0.f;
#pragma unroll
  for (int i = 0; i < EIN_; ++i) ej += eas[i] * we[i * HID_ + j];
  int s = src[e], d = dst[e];
  float t = Q[(size_t)d * HID_ + j] * (Kb[(size_t)s * HID_ + j] + ej);
  t = wave_sum32(t);
  int lane = j & 31, hh = j >> 5;
  if (lane == 0) {
    float lg = t * 0.17677669529663687f;   // 1/sqrt(32)
    logits[(size_t)e * H_ + hh] = lg;
    atomicMax(&mU[(size_t)d * H_ + hh], enc_f32(lg));
  }
}

// pass 2: p = exp(logit - m), segment sum
__global__ void attn_expsum(float* __restrict__ logits, const unsigned* __restrict__ mU,
                            const int* __restrict__ dst, float* __restrict__ sB) {
  int idx = blockIdx.x * blockDim.x + threadIdx.x;
  if (idx >= E_ * H_) return;
  int e = idx >> 2, hh = idx & 3;
  int d = dst[e];
  float p = __expf(logits[idx] - dec_f32(mU[(size_t)d * H_ + hh]));
  logits[idx] = p;
  atomicAdd(&sB[(size_t)d * H_ + hh], p);
}

// pass 3: out[dst] += alpha * (V[src] + e)   (out pre-seeded with skip GEMM)
__global__ void attn_scatter(const float* __restrict__ V, const float* __restrict__ we,
                             const float* __restrict__ ea, const int* __restrict__ src,
                             const int* __restrict__ dst, const float* __restrict__ p,
                             const float* __restrict__ sB, float* __restrict__ outb) {
  int e = blockIdx.x, j = threadIdx.x;
  __shared__ float eas[EIN_];
  if (j < EIN_) eas[j] = ea[(size_t)e * EIN_ + j];
  __syncthreads();
  float ej = 0.f;
#pragma unroll
  for (int i = 0; i < EIN_; ++i) ej += eas[i] * we[i * HID_ + j];
  int s = src[e], d = dst[e], hh = j >> 5;
  float alpha = p[(size_t)e * H_ + hh] / (sB[(size_t)d * H_ + hh] + 1e-16f);
  atomicAdd(&outb[(size_t)d * HID_ + j], alpha * (V[(size_t)s * HID_ + j] + ej));
}

// layernorm over 128 + relu -> h (f32) and hh (f16)
__global__ void ln_relu(const float* __restrict__ outb, const float* __restrict__ g,
                        const float* __restrict__ b, float* __restrict__ h,
                        _Float16* __restrict__ hh) {
  int n = blockIdx.x, j = threadIdx.x;   // 128
  float v = outb[(size_t)n * HID_ + j];
  __shared__ float r1[4], r2[4];
  int wid = j >> 5, lane = j & 31;
  float t = wave_sum32(v);
  if (lane == 0) r1[wid] = t;
  __syncthreads();
  float mu = (r1[0] + r1[1] + r1[2] + r1[3]) * (1.f / 128.f);
  float dv = v - mu;
  float t2 = wave_sum32(dv * dv);
  if (lane == 0) r2[wid] = t2;
  __syncthreads();
  float var = (r2[0] + r2[1] + r2[2] + r2[3]) * (1.f / 128.f);
  float y = fmaxf(dv * rsqrtf(var + 1e-5f) * g[j] + b[j], 0.f);
  h[(size_t)n * HID_ + j] = y;
  hh[(size_t)n * HID_ + j] = (_Float16)y;
}

// per-graph mean pool + readout
__global__ void readout_k(const float* __restrict__ h, const float* __restrict__ w,
                          const float* __restrict__ b, float* __restrict__ outp) {
  int g = blockIdx.x, j = threadIdx.x;   // 128
  float s = 0.f;
  for (int l = 0; l < NPER_; ++l) s += h[(size_t)(g * NPER_ + l) * HID_ + j];
  float part = (s * (1.f / 512.f)) * w[j];
  part = wave_sum32(part);
  __shared__ float r[4];
  int wid = j >> 5, lane = j & 31;
  if (lane == 0) r[wid] = part;
  __syncthreads();
  if (j == 0) outp[g] = r[0] + r[1] + r[2] + r[3] + b[0];
}

// ---------------- launch ----------------
extern "C" void kernel_launch(void* const* d_in, const int* in_sizes, int n_in,
                              void* d_out, int out_size, void* d_ws, size_t ws_size,
                              hipStream_t stream) {
  (void)in_sizes; (void)n_in; (void)out_size; (void)ws_size;
  const float* x        = (const float*)d_in[0];
  const float* eattr    = (const float*)d_in[1];
  const int*   esrc     = (const int*)  d_in[2];
  const int*   edst     = (const int*)  d_in[3];
  const float* emlp_w1  = (const float*)d_in[4];
  const float* emlp_b1  = (const float*)d_in[5];
  const float* emlp_w2  = (const float*)d_in[6];
  const float* emlp_b2  = (const float*)d_in[7];
  const float* lr_w     = (const float*)d_in[8];
  const float* lr_b     = (const float*)d_in[9];
  const float* ro_w     = (const float*)d_in[10];
  const float* ro_b     = (const float*)d_in[11];
  float* outp = (float*)d_out;

  // workspace layout (256B aligned slices)
  char* ws = (char*)d_ws;
  size_t off = 0;
  auto alloc = [&](size_t bytes) { void* p = ws + off; off += (bytes + 255) & ~(size_t)255; return p; };
  const size_t RELEM = (size_t)G_ * NPER_ * NPER_;   // 8388608
  float*     Ff   = (float*)    alloc((size_t)E_ * 4);
  float*     R    = (float*)    alloc(RELEM * 4);
  _Float16*  Rh   = (_Float16*) alloc(RELEM * 2);
  _Float16*  RhT  = (_Float16*) alloc(RELEM * 2);
  float*     P    = (float*)    alloc(RELEM * 4);
  float*     h    = (float*)    alloc((size_t)N_ * HID_ * 4);
  _Float16*  hh   = (_Float16*) alloc((size_t)N_ * HID_ * 2);
  float*     Q    = (float*)    alloc((size_t)N_ * HID_ * 4);
  float*     Kb   = (float*)    alloc((size_t)N_ * HID_ * 4);
  float*     V    = (float*)    alloc((size_t)N_ * HID_ * 4);
  float*     OUT  = (float*)    alloc((size_t)N_ * HID_ * 4);
  float*     LOG  = (float*)    alloc((size_t)E_ * H_ * 4);
  unsigned*  MU   = (unsigned*) alloc((size_t)N_ * H_ * 4);
  float*     SB   = (float*)    alloc((size_t)N_ * H_ * 4);
  _Float16*  wqT  = (_Float16*) alloc((size_t)HID_ * HID_ * 2);
  _Float16*  wkT  = (_Float16*) alloc((size_t)HID_ * HID_ * 2);
  _Float16*  wvT  = (_Float16*) alloc((size_t)HID_ * HID_ * 2);
  _Float16*  wsT  = (_Float16*) alloc((size_t)HID_ * HID_ * 2);

  // ---- adaptive edge strengths, dense per-graph kernel ----
  edge_mlp<<<E_ / 256, 256, 0, stream>>>(eattr, emlp_w1, emlp_b1, emlp_w2, emlp_b2, Ff);
  zero_u32<<<(int)(RELEM / 256), 256, 0, stream>>>((unsigned*)R, (int)RELEM);
  scatter_R<<<E_ / 256, 256, 0, stream>>>(esrc, edst, Ff, R);
  colnorm_R<<<(G_ * NPER_) / 256, 256, 0, stream>>>(R, Rh);
  transpose_h16<<<dim3(16, 16, G_), dim3(32, 8), 0, stream>>>(Rh, RhT);
  // P = R @ R  (batched WMMA, B fed via RhT rows); (512/32)^2 = 256 tiles/graph
  wmma_gemm_kernel<<<dim3(32, G_), 256, 0, stream>>>(
      Rh, RhT, nullptr, P, NPER_, NPER_, NPER_, NPER_, NPER_, NPER_,
      (long long)NPER_ * NPER_, (long long)NPER_ * NPER_, (long long)NPER_ * NPER_);
  build_h0<<<N_ / 8, 256, 0, stream>>>(R, RhT, P, x, lr_w, lr_b, h, hh);

  // ---- two TransformerConv layers ----
  for (int L = 0; L < 2; ++L) {
    int base = 12 + L * 11;
    const float* wq = (const float*)d_in[base + 0];
    const float* bq = (const float*)d_in[base + 1];
    const float* wk = (const float*)d_in[base + 2];
    const float* bk = (const float*)d_in[base + 3];
    const float* wv = (const float*)d_in[base + 4];
    const float* bv = (const float*)d_in[base + 5];
    const float* we = (const float*)d_in[base + 6];
    const float* wskip = (const float*)d_in[base + 7];
    const float* bskip = (const float*)d_in[base + 8];
    const float* lng = (const float*)d_in[base + 9];
    const float* lnb = (const float*)d_in[base + 10];
    const int Kd = (L == 0) ? (IN_ + LR_) : HID_;   // 96 or 128 (mult of 32)

    wT_f16<<<64, 256, 0, stream>>>(wq, wqT, Kd);
    wT_f16<<<64, 256, 0, stream>>>(wk, wkT, Kd);
    wT_f16<<<64, 256, 0, stream>>>(wv, wvT, Kd);
    wT_f16<<<64, 256, 0, stream>>>(wskip, wsT, Kd);

    // Q/K/V and skip-seeded OUT: [16384 x Kd] x [Kd x 128]
    // (16384/32)*(128/32) = 2048 tiles, 8 waves/block -> 256 blocks
    wmma_gemm_kernel<<<dim3(256, 1), 256, 0, stream>>>(hh, wqT, bq, Q,
        N_, HID_, Kd, HID_, HID_, HID_, 0, 0, 0);
    wmma_gemm_kernel<<<dim3(256, 1), 256, 0, stream>>>(hh, wkT, bk, Kb,
        N_, HID_, Kd, HID_, HID_, HID_, 0, 0, 0);
    wmma_gemm_kernel<<<dim3(256, 1), 256, 0, stream>>>(hh, wvT, bv, V,
        N_, HID_, Kd, HID_, HID_, HID_, 0, 0, 0);
    wmma_gemm_kernel<<<dim3(256, 1), 256, 0, stream>>>(hh, wsT, bskip, OUT,
        N_, HID_, Kd, HID_, HID_, HID_, 0, 0, 0);

    zero_u32<<<(N_ * H_) / 256, 256, 0, stream>>>(MU, N_ * H_);
    zero_u32<<<(N_ * H_) / 256, 256, 0, stream>>>((unsigned*)SB, N_ * H_);
    attn_logits<<<E_, 128, 0, stream>>>(Q, Kb, we, eattr, esrc, edst, LOG, MU);
    attn_expsum<<<(E_ * H_) / 256, 256, 0, stream>>>(LOG, MU, edst, SB);
    attn_scatter<<<E_, 128, 0, stream>>>(V, we, eattr, esrc, edst, LOG, SB, OUT);
    ln_relu<<<N_, 128, 0, stream>>>(OUT, lng, lnb, h, hh);
  }

  readout_k<<<G_, 128, 0, stream>>>(h, ro_w, ro_b, outp);
}
